// RNN_9165460210264
// MI455X (gfx1250) — compile-verified, matrix operand order checked by reading, not run on hardware
//
#include <hip/hip_runtime.h>

typedef __attribute__((ext_vector_type(16))) __bf16 v16bf;
typedef __attribute__((ext_vector_type(8)))  float  v8f;

#define T_LEN 512
#define B_SZ  128
#define HID   256
#define H4    1024
#define BWG   32            // batch rows per workgroup (two 16-row M tiles share B frags)
#define LDSK  264           // padded h-row stride (halves): conflict-free b128 reads
#define XSTR  520           // padded staged-input row stride (halves)

// ---------- helpers ----------
__device__ __forceinline__ unsigned short f2bf(float f) {
    unsigned int u = __float_as_uint(f);
    unsigned int r = (u + 0x7FFFu + ((u >> 16) & 1u)) >> 16;   // round-nearest-even
    return (unsigned short)r;
}
__device__ __forceinline__ float sigf(float x) { return 1.f / (1.f + __expf(-x)); }
__device__ __forceinline__ float tanha(float x) {
    x = fminf(fmaxf(x, -15.f), 15.f);
    float e = __expf(2.f * x);
    return (e - 1.f) / (e + 1.f);
}
// A-fragment (16x32 bf16) from a row-major [16][stride] tile: lane row = lane&15,
// lanes<16 take K {kb..kb+7, kb+16..kb+23}, lanes>=16 same with +8 (kb passed pre-offset).
__device__ __forceinline__ v16bf load_frag(const unsigned short* p) {
    union { uint4 q[2]; v16bf v; } u;
    u.q[0] = *(const uint4*)(p);
    u.q[1] = *(const uint4*)(p + 16);
    return u.v;
}
// B-fragment from pre-swizzled weights: 512B-coalesced per chunk
__device__ __forceinline__ v16bf load_wfrag(const unsigned short* p, int lane) {
    union { uint4 q[2]; v16bf v; } u;
    u.q[0] = *(const uint4*)(p + lane * 8);
    u.q[1] = *(const uint4*)(p + 256 + lane * 8);
    return u.v;
}
#define NO_HOIST(p) asm volatile("" : "+v"(p))

// ---------- weight pre-swizzle: f32 [2][1024][K] -> bf16 fragment order ----------
// dst[(((((dir*16+s)*KT + kt)*4 + q)*2 + c)*32 + lane)*8 + e]
//   = bf16( src[dir][q*256 + s*16 + (lane&15)][kt*32 + c*16 + (lane>>4)*8 + e] )
__global__ void pack_w_kernel(const float* __restrict__ src,
                              unsigned short* __restrict__ dst, int KT) {
    int i = blockIdx.x * blockDim.x + threadIdx.x;
    int total = KT * 65536;
    if (i >= total) return;
    int e    = i & 7;
    int lane = (i >> 3) & 31;
    int c    = (i >> 8) & 1;
    int q    = (i >> 9) & 3;
    int kt   = (i >> 11) % KT;
    int sd   = (i >> 11) / KT;
    int s    = sd & 15;
    int dir  = sd >> 4;
    int n = q * HID + s * 16 + (lane & 15);
    int k = kt * 32 + c * 16 + (lane >> 4) * 8 + e;
    int K = KT * 32;
    dst[i] = f2bf(src[((size_t)dir * H4 + n) * K + k]);
}

// ---------- layer 0: persistent bidirectional LSTM scan (input size 4) ----------
__global__ __launch_bounds__(512, 1)
void lstm_layer0(const float* __restrict__ x,            // [B][T][4]
                 const float* __restrict__ wi0,          // [2][1024][4]
                 const float* __restrict__ bi0,          // [2][1024]
                 const float* __restrict__ bh0,          // [2][1024]
                 const unsigned short* __restrict__ wp0, // packed wh0, KT=8
                 unsigned short* __restrict__ h0cat)     // out [T][B][512] bf16 (fwd|bwd)
{
    const int dir  = blockIdx.y;
    const int mb   = blockIdx.x * BWG;
    const int tid  = threadIdx.x;
    const int s    = tid >> 5;          // wave id == hidden slice (16 cols)
    const int lane = tid & 31;
    const int ln   = lane & 15;
    const int hi   = lane >> 4;

    __shared__ unsigned short hb[2][BWG * LDSK];
    for (int i = tid; i < 2 * BWG * LDSK; i += 512) ((unsigned short*)hb)[i] = 0;
    __syncthreads();

    const unsigned short* wq = wp0 + (size_t)(dir * 16 + s) * 8 * 2048;

    float  bias[4];
    float4 wrow[4];
    #pragma unroll
    for (int q = 0; q < 4; ++q) {                       // gate order: i,f,g,o
        int n = q * HID + s * 16 + ln;
        bias[q] = bi0[dir * H4 + n] + bh0[dir * H4 + n];
        const float* wr = wi0 + ((size_t)dir * H4 + n) * 4;
        wrow[q] = make_float4(wr[0], wr[1], wr[2], wr[3]);
    }

    v8f c[2] = {{}, {}};
    for (int tt = 0; tt < T_LEN; ++tt) {
        const int t = dir ? (T_LEN - 1 - tt) : tt;
        const int p = tt & 1;
        const unsigned short* wqo = wq;
        NO_HOIST(wqo);                                  // keep weight loads inside the loop

        // input contribution (K=4): VALU FMAs
        v8f acc[2][4];
        #pragma unroll
        for (int m = 0; m < 2; ++m) {
            float4 xv[8];
            #pragma unroll
            for (int j = 0; j < 8; ++j) {
                int row = mb + m * 16 + j + 8 * hi;
                xv[j] = *(const float4*)(x + ((size_t)row * T_LEN + t) * 4);
            }
            #pragma unroll
            for (int q = 0; q < 4; ++q)
                #pragma unroll
                for (int j = 0; j < 8; ++j)
                    acc[m][q][j] = bias[q] + wrow[q].x * xv[j].x + wrow[q].y * xv[j].y
                                           + wrow[q].z * xv[j].z + wrow[q].w * xv[j].w;
        }

        // recurrent GEMM: h(t-1)[32x256] @ wh^T, K tiled by 32; B frags shared by both M tiles
        #pragma unroll
        for (int kt = 0; kt < 8; ++kt) {
            const int kb = kt * 32 + hi * 8;
            v16bf a0 = load_frag(&hb[p][ln * LDSK + kb]);
            v16bf a1 = load_frag(&hb[p][(16 + ln) * LDSK + kb]);
            #pragma unroll
            for (int q = 0; q < 4; ++q) {
                v16bf b = load_wfrag(wqo + kt * 2048 + q * 512, lane);
                acc[0][q] = __builtin_amdgcn_wmma_f32_16x16x32_bf16(
                    false, a0, false, b, (short)0, acc[0][q], false, false);
                acc[1][q] = __builtin_amdgcn_wmma_f32_16x16x32_bf16(
                    false, a1, false, b, (short)0, acc[1][q], false, false);
            }
        }

        // LSTM cell update (i/f/g/o tiles share lane mapping) + publish h(t)
        #pragma unroll
        for (int m = 0; m < 2; ++m)
            #pragma unroll
            for (int j = 0; j < 8; ++j) {
                float ci = sigf(acc[m][1][j]) * c[m][j]
                         + sigf(acc[m][0][j]) * tanha(acc[m][2][j]);
                c[m][j] = ci;
                float hv = sigf(acc[m][3][j]) * tanha(ci);
                int r = m * 16 + j + 8 * hi;
                unsigned short hv16 = f2bf(hv);
                hb[p ^ 1][r * LDSK + s * 16 + ln] = hv16;
                h0cat[((size_t)t * B_SZ + mb + r) * (2 * HID) + dir * HID + s * 16 + ln] = hv16;
            }
        __syncthreads();                                // one barrier/step (double buffer)
    }
}

// ---------- layer 1: persistent bidirectional LSTM scan (input = 512 from h0cat) ----------
__global__ __launch_bounds__(512, 1)
void lstm_layer1(const unsigned short* __restrict__ h0cat, // [T][B][512] bf16
                 const float* __restrict__ bi1,
                 const float* __restrict__ bh1,
                 const unsigned short* __restrict__ wpi1,  // packed w_ih1, KT=16
                 const unsigned short* __restrict__ wph1,  // packed w_hh1, KT=8
                 float* __restrict__ hsum)                 // out [B][512] f32 (sum over t)
{
    const int dir  = blockIdx.y;
    const int mb   = blockIdx.x * BWG;
    const int tid  = threadIdx.x;
    const int s    = tid >> 5;
    const int lane = tid & 31;
    const int ln   = lane & 15;
    const int hi   = lane >> 4;

    __shared__ unsigned short hb[2][BWG * LDSK];
    __shared__ unsigned short xst[2][BWG * XSTR];          // staged input tiles (async)
    for (int i = tid; i < 2 * BWG * LDSK; i += 512) ((unsigned short*)hb)[i] = 0;

    const unsigned short* wqi = wpi1 + (size_t)(dir * 16 + s) * 16 * 2048;
    const unsigned short* wqh = wph1 + (size_t)(dir * 16 + s) * 8 * 2048;

    float bias[4];
    #pragma unroll
    for (int q = 0; q < 4; ++q) {
        int n = q * HID + s * 16 + ln;
        bias[q] = bi1[dir * H4 + n] + bh1[dir * H4 + n];
    }

    // async-stage thread mapping: each thread copies 32 halves of one row
    const int sr = tid >> 4;            // row 0..31
    const int sc = (tid & 15) * 32;     // col base

    // prologue: stage tile for step 0 into buffer 0
    {
        const int t0 = dir ? (T_LEN - 1) : 0;
        const unsigned short* g = h0cat + ((size_t)t0 * B_SZ + mb + sr) * (2 * HID) + sc;
        unsigned int l = (unsigned int)(size_t)&xst[0][sr * XSTR + sc];
        #pragma unroll
        for (int o = 0; o < 4; ++o) {
            unsigned long long ga = (unsigned long long)(g + o * 8);
            unsigned int la = l + o * 16;
            asm volatile("global_load_async_to_lds_b128 %0, %1, off" :: "v"(la), "v"(ga) : "memory");
        }
        asm volatile("s_wait_asynccnt 0x0" ::: "memory");
    }
    __syncthreads();

    v8f c[2]  = {{}, {}};
    v8f hs[2] = {{}, {}};                                  // running sum over t (mean pool)
    for (int tt = 0; tt < T_LEN; ++tt) {
        const int t = dir ? (T_LEN - 1 - tt) : tt;
        const int p = tt & 1;

        // kick off async staging of the NEXT step's input tile (overlaps WMMAs below)
        if (tt + 1 < T_LEN) {
            const int tn = dir ? (T_LEN - 2 - tt) : (tt + 1);
            const unsigned short* g = h0cat + ((size_t)tn * B_SZ + mb + sr) * (2 * HID) + sc;
            unsigned int l = (unsigned int)(size_t)&xst[p ^ 1][sr * XSTR + sc];
            #pragma unroll
            for (int o = 0; o < 4; ++o) {
                unsigned long long ga = (unsigned long long)(g + o * 8);
                unsigned int la = l + o * 16;
                asm volatile("global_load_async_to_lds_b128 %0, %1, off" :: "v"(la), "v"(ga) : "memory");
            }
        }

        const unsigned short* wio = wqi;  NO_HOIST(wio);
        const unsigned short* who = wqh;  NO_HOIST(who);

        v8f acc[2][4];
        #pragma unroll
        for (int m = 0; m < 2; ++m)
            #pragma unroll
            for (int q = 0; q < 4; ++q)
                #pragma unroll
                for (int j = 0; j < 8; ++j) acc[m][q][j] = bias[q];

        // input GEMM, K=512, A from staged LDS tile
        #pragma unroll
        for (int kt = 0; kt < 16; ++kt) {
            const int kb = kt * 32 + hi * 8;
            v16bf a0 = load_frag(&xst[p][ln * XSTR + kb]);
            v16bf a1 = load_frag(&xst[p][(16 + ln) * XSTR + kb]);
            #pragma unroll
            for (int q = 0; q < 4; ++q) {
                v16bf b = load_wfrag(wio + kt * 2048 + q * 512, lane);
                acc[0][q] = __builtin_amdgcn_wmma_f32_16x16x32_bf16(
                    false, a0, false, b, (short)0, acc[0][q], false, false);
                acc[1][q] = __builtin_amdgcn_wmma_f32_16x16x32_bf16(
                    false, a1, false, b, (short)0, acc[1][q], false, false);
            }
        }
        // recurrent GEMM, K=256, A from h double buffer
        #pragma unroll
        for (int kt = 0; kt < 8; ++kt) {
            const int kb = kt * 32 + hi * 8;
            v16bf a0 = load_frag(&hb[p][ln * LDSK + kb]);
            v16bf a1 = load_frag(&hb[p][(16 + ln) * LDSK + kb]);
            #pragma unroll
            for (int q = 0; q < 4; ++q) {
                v16bf b = load_wfrag(who + kt * 2048 + q * 512, lane);
                acc[0][q] = __builtin_amdgcn_wmma_f32_16x16x32_bf16(
                    false, a0, false, b, (short)0, acc[0][q], false, false);
                acc[1][q] = __builtin_amdgcn_wmma_f32_16x16x32_bf16(
                    false, a1, false, b, (short)0, acc[1][q], false, false);
            }
        }

        #pragma unroll
        for (int m = 0; m < 2; ++m)
            #pragma unroll
            for (int j = 0; j < 8; ++j) {
                float ci = sigf(acc[m][1][j]) * c[m][j]
                         + sigf(acc[m][0][j]) * tanha(acc[m][2][j]);
                c[m][j] = ci;
                float hv = sigf(acc[m][3][j]) * tanha(ci);
                hs[m][j] += hv;
                int r = m * 16 + j + 8 * hi;
                hb[p ^ 1][r * LDSK + s * 16 + ln] = f2bf(hv);
            }

        asm volatile("s_wait_asynccnt 0x0" ::: "memory");  // next-step stage complete
        __syncthreads();
    }

    #pragma unroll
    for (int m = 0; m < 2; ++m)
        #pragma unroll
        for (int j = 0; j < 8; ++j) {
            int r = m * 16 + j + 8 * hi;
            hsum[(size_t)(mb + r) * (2 * HID) + dir * HID + s * 16 + ln] = hs[m][j];
        }
}

// ---------- mean-pool + FC ----------
__global__ void fc_kernel(const float* __restrict__ hsum,
                          const float* __restrict__ fcw,   // [2][512]
                          const float* __restrict__ fcb,   // [2]
                          float* __restrict__ out)         // [128][2]
{
    int id = blockIdx.x * blockDim.x + threadIdx.x;
    if (id >= B_SZ * 2) return;
    int b = id >> 1, o = id & 1;
    float acc = 0.f;
    for (int j = 0; j < 2 * HID; ++j)
        acc += hsum[(size_t)b * (2 * HID) + j] * fcw[o * (2 * HID) + j];
    out[b * 2 + o] = acc * (1.f / (float)T_LEN) + fcb[o];
}

extern "C" void kernel_launch(void* const* d_in, const int* in_sizes, int n_in,
                              void* d_out, int out_size, void* d_ws, size_t ws_size,
                              hipStream_t stream) {
    (void)in_sizes; (void)n_in; (void)out_size; (void)ws_size;
    const float* x     = (const float*)d_in[0];
    const float* w_ih0 = (const float*)d_in[1];
    const float* w_hh0 = (const float*)d_in[2];
    const float* b_ih0 = (const float*)d_in[3];
    const float* b_hh0 = (const float*)d_in[4];
    const float* w_ih1 = (const float*)d_in[5];
    const float* w_hh1 = (const float*)d_in[6];
    const float* b_ih1 = (const float*)d_in[7];
    const float* b_hh1 = (const float*)d_in[8];
    const float* fc_w  = (const float*)d_in[9];
    const float* fc_b  = (const float*)d_in[10];
    float* out = (float*)d_out;

    const int N_WH  = 8  * 65536;          // packed wh (KT=8):  524288 halves
    const int N_WI1 = 16 * 65536;          // packed w_ih1 (KT=16): 1048576 halves
    unsigned short* wp0   = (unsigned short*)d_ws;
    unsigned short* wph1  = wp0  + N_WH;
    unsigned short* wpi1  = wph1 + N_WH;
    unsigned short* h0cat = wpi1 + N_WI1;                      // T*B*512 halves = 64 MiB
    float*          hsum  = (float*)(h0cat + (size_t)T_LEN * B_SZ * 2 * HID);

    pack_w_kernel<<<(N_WH  + 255) / 256, 256, 0, stream>>>(w_hh0, wp0,  8);
    pack_w_kernel<<<(N_WH  + 255) / 256, 256, 0, stream>>>(w_hh1, wph1, 8);
    pack_w_kernel<<<(N_WI1 + 255) / 256, 256, 0, stream>>>(w_ih1, wpi1, 16);

    dim3 grid(B_SZ / BWG, 2);              // 4 batch groups x 2 directions
    lstm_layer0<<<grid, 512, 0, stream>>>(x, w_ih0, b_ih0, b_hh0, wp0, h0cat);
    lstm_layer1<<<grid, 512, 0, stream>>>(h0cat, b_ih1, b_hh1, wpi1, wph1, hsum);

    fc_kernel<<<1, 256, 0, stream>>>(hsum, fc_w, fc_b, out);
}
